// MetaDKF_62045097558102
// MI455X (gfx1250) — compile-verified, matrix operand order checked by reading, not run on hardware
//
#include <hip/hip_runtime.h>

typedef _Float16 half_t;
typedef __attribute__((ext_vector_type(16))) _Float16 v16h;
typedef __attribute__((ext_vector_type(8)))  float    v8f;

#define BM 128
#define BN 64
#define BKQ 32
#define BKP (BKQ + 8)   // +8 halves: keeps 16B alignment for async B128, spreads banks
#define WAVES 8

#define ACT_NONE 0
#define ACT_RELU 1
#define ACT_SIGM 2
#define ACT_TANH 3
#define ACT_HALFSUM_TANH 4

struct ConvGeom {
  int Cin, Hin, Win, Hout, Wout, KH, KW, stride, pad, deconv, chanMajor;
};

// ---------------- fast math helpers ----------------

__device__ __forceinline__ float fast_rcp(float x) { return __builtin_amdgcn_rcpf(x); }
__device__ __forceinline__ float fast_sigm(float x) {
  return fast_rcp(1.0f + __expf(-x));                 // 1 v_exp + 1 v_rcp
}
__device__ __forceinline__ float fast_tanh(float x) {
  return 1.0f - 2.0f * fast_rcp(1.0f + __expf(2.0f * x));
}

__device__ __forceinline__ unsigned hash_u32(unsigned x) {
  x ^= x >> 16; x *= 0x7feb352dU; x ^= x >> 15; x *= 0x846ca68bU; x ^= x >> 16;
  return x;
}

__device__ __forceinline__ float randn_at(unsigned seed, unsigned idx) {
  unsigned h1 = hash_u32(seed ^ (idx * 2u + 0x9E3779B9u));
  unsigned h2 = hash_u32(seed + 0x85EBCA6Bu + idx * 2u + 1u);
  float u1 = ((h1 >> 8) + 1u) * (1.0f / 16777217.0f);
  float u2 = (h2 >> 8) * (1.0f / 16777216.0f);
  return sqrtf(fmaxf(-2.0f * __logf(u1), 0.0f)) * __cosf(6.28318530718f * u2);
}

__device__ __forceinline__ float act_apply(float v, int act, float post) {
  if (act == ACT_RELU) return fmaxf(v, 0.0f);
  if (act == ACT_SIGM) return fast_sigm(v);
  if (act == ACT_TANH) return fast_tanh(v);
  if (act == ACT_HALFSUM_TANH) return 0.5f * (fast_tanh(v) + post);
  return v;
}

// ---------------- async global->LDS copy (CDNA5, ASYNCcnt-tracked) ----------------

__device__ __forceinline__ void async_copy_b128(unsigned ldsAddr, const void* gaddr) {
  asm volatile("global_load_async_to_lds_b128 %0, %1, off"
               :: "v"(ldsAddr), "v"((unsigned long long)(uintptr_t)gaddr)
               : "memory");
}
__device__ __forceinline__ void async_wait0() {
  asm volatile("s_wait_asynccnt 0" ::: "memory");
}

// A-operand fetch for the generic tiled GEMM (sync staging path).
// MODE 0: dense row-major [M,K]
// MODE 1: dense with NHWC->NCHW k-permutation (k = c*SP + s ; addr = m*K + s*CH + c)
// MODE 2: implicit im2col for conv (stride/pad) or deconv (lhs_dilation=2, stride 1)
template<int MODE, bool AH>
__device__ __forceinline__ float fetchA(const void* A, long m, int k, int Kreal,
                                        int permCH, int permSP, const ConvGeom& g) {
  long idx;
  if constexpr (MODE == 0) {
    idx = m * (long)Kreal + k;
  } else if constexpr (MODE == 1) {
    int c = k / permSP;
    int s = k - c * permSP;
    idx = m * (long)Kreal + (long)s * permCH + c;
  } else {
    int hw = g.Hout * g.Wout;
    long fr = m / hw;
    int rem = (int)(m - fr * (long)hw);
    int oy = rem / g.Wout, ox = rem - oy * g.Wout;
    int khw = g.KH * g.KW;
    int ic = k / khw;
    int r2 = k - ic * khw;
    int ky = r2 / g.KW, kx = r2 - ky * g.KW;
    int y, xx;
    bool ok;
    if (!g.deconv) {
      y = oy * g.stride - g.pad + ky;
      xx = ox * g.stride - g.pad + kx;
      ok = (y >= 0 && y < g.Hin && xx >= 0 && xx < g.Win);
    } else {
      int yd = oy - g.pad + ky;
      int xd = ox - g.pad + kx;
      ok = (yd >= 0 && xd >= 0 && yd < 2 * g.Hin - 1 && xd < 2 * g.Win - 1 &&
            ((yd & 1) == 0) && ((xd & 1) == 0));
      y = yd >> 1; xx = xd >> 1;
    }
    if (!ok) return 0.0f;
    if (g.chanMajor) idx = ((fr * (long)g.Cin + ic) * g.Hin + y) * g.Win + xx;
    else             idx = ((fr * (long)g.Hin + y) * g.Win + xx) * g.Cin + ic;
  }
  if constexpr (AH) return (float)((const half_t*)A)[idx];
  else              return ((const float*)A)[idx];
}

// ---------------- generic tiled WMMA GEMM ----------------
// C[M,N] = act(A[M,K] @ W[N,K]^T + bias), tiles 128x64x32, 8 waves/block.
// Double-buffered LDS; ASYNCB path uses GLOBAL_LOAD_ASYNC_TO_LDS_B128 DMA
// (requires f16 A, M%128==0, N%64==0, K%32==0 — guaranteed at call sites).
template<int MODE, bool AH, bool OH, bool ASYNCB>
__global__ __launch_bounds__(256) void gemm_tiled(
    const void* __restrict__ A, const half_t* __restrict__ W,
    const float* __restrict__ bias, void* __restrict__ Cout,
    long M, int Nrl, int Kreal, int act, int permCH, int permSP, ConvGeom g) {
  __shared__ half_t As[2][BM][BKP];
  __shared__ half_t Bs[2][BN][BKP];
  const int tid = threadIdx.x;
  const int wid = tid >> 5, lane = tid & 31;
  const int lm = lane & 15, lg = lane >> 4;
  const long mBase = (long)blockIdx.x * BM;
  const int nBase = blockIdx.y * BN;
  const int nk = (Kreal + BKQ - 1) / BKQ;

  float aReg[16];
  float bReg[8];

  auto issue_async = [&](int buf, int kb) {
    // A tile: 128 rows x 32 halves = 512 B128 chunks, 2 per thread
#pragma unroll
    for (int i = 0; i < 2; ++i) {
      int c0 = tid * 2 + i;
      int r = c0 >> 2, o = (c0 & 3) * 8;
      const half_t* ga = (const half_t*)A + (mBase + r) * (long)Kreal + kb + o;
      async_copy_b128((unsigned)(uintptr_t)&As[buf][r][o], ga);
    }
    // B tile: 64 rows x 32 halves = 256 B128 chunks, 1 per thread
    {
      int r = tid >> 2, o = (tid & 3) * 8;
      const half_t* gb = W + (long)(nBase + r) * Kreal + kb + o;
      async_copy_b128((unsigned)(uintptr_t)&Bs[buf][r][o], gb);
    }
  };

  auto fetch_tiles = [&](int kb) {
#pragma unroll
    for (int i = 0; i < 16; ++i) {
      int e = tid + i * 256;
      int r = e >> 5, c = e & 31;
      long m = mBase + r;
      int k = kb + c;
      float v = 0.0f;
      if (m < M && k < Kreal) v = fetchA<MODE, AH>(A, m, k, Kreal, permCH, permSP, g);
      aReg[i] = v;
    }
#pragma unroll
    for (int i = 0; i < 8; ++i) {
      int e = tid + i * 256;
      int r = e >> 5, c = e & 31;
      int n = nBase + r;
      int k = kb + c;
      float v = 0.0f;
      if (n < Nrl && k < Kreal) v = (float)W[(long)n * Kreal + k];
      bReg[i] = v;
    }
  };
  auto store_tiles = [&](int buf) {
#pragma unroll
    for (int i = 0; i < 16; ++i) {
      int e = tid + i * 256;
      As[buf][e >> 5][e & 31] = (half_t)aReg[i];
    }
#pragma unroll
    for (int i = 0; i < 8; ++i) {
      int e = tid + i * 256;
      Bs[buf][e >> 5][e & 31] = (half_t)bReg[i];
    }
  };

  v8f acc[4] = {};

  if constexpr (ASYNCB) {
    issue_async(0, 0);
    async_wait0();
  } else {
    fetch_tiles(0);
    store_tiles(0);
  }
  __syncthreads();

  int buf = 0;
  for (int t = 0; t < nk; ++t) {
    const int kn = (t + 1) * BKQ;
    if (t + 1 < nk) {
      if constexpr (ASYNCB) issue_async(buf ^ 1, kn);  // DMA next tile
      else fetch_tiles(kn);                            // loads in flight during wmma
    }

    // A fragment: lane=M row, K pairs per ISA 16-bit A layout
    v16h a;
    int arow = wid * 16 + lm;
#pragma unroll
    for (int r = 0; r < 8; ++r) {
      int kp = (r < 4 ? 2 * r : 16 + 2 * (r - 4)) + 8 * lg;
      a[2 * r] = As[buf][arow][kp];
      a[2 * r + 1] = As[buf][arow][kp + 1];
    }
#pragma unroll
    for (int c = 0; c < 4; ++c) {
      v16h b;
      int brow = c * 16 + lm;
#pragma unroll
      for (int r = 0; r < 8; ++r) {
        int kp = (r < 4 ? 2 * r : 16 + 2 * (r - 4)) + 8 * lg;
        b[2 * r] = Bs[buf][brow][kp];
        b[2 * r + 1] = Bs[buf][brow][kp + 1];
      }
      acc[c] = __builtin_amdgcn_wmma_f32_16x16x32_f16(false, a, false, b,
                                                      (short)0, acc[c], false, false);
    }

    if (t + 1 < nk) {
      if constexpr (ASYNCB) async_wait0();
      else store_tiles(buf ^ 1);
      __syncthreads();
      buf ^= 1;
    }
  }

  // epilogue: C/D layout — vgpr e -> M = lg*8 + e, lane lm -> N
#pragma unroll
  for (int c = 0; c < 4; ++c) {
    int n = nBase + c * 16 + lm;
    if (n >= Nrl) continue;
    float bb = bias ? bias[n] : 0.0f;
#pragma unroll
    for (int e = 0; e < 8; ++e) {
      long m = mBase + wid * 16 + lg * 8 + e;
      if (m >= M) continue;
      float v = acc[c][e] + bb;
      if (act == ACT_RELU) v = fmaxf(v, 0.0f);
      if constexpr (OH) ((half_t*)Cout)[m * (long)Nrl + n] = (half_t)v;
      else              ((float*)Cout)[m * (long)Nrl + n] = v;
    }
  }
}

// ---------------- block-level GEMM for persistent kernels ----------------
// M fixed = 64, A f16 in LDS [64][lda], W f16 in global [N,K] (L2-resident).
// Register-double-buffered K loop: loads for chunk k+1 issue before the WMMA
// on chunk k so memory latency overlaps the matrix op (partial s_wait counts).
__device__ void block_gemm_lds(const half_t* A, int lda, int K,
                               const half_t* __restrict__ W, const float* __restrict__ bias,
                               int N, void* out, int ldo, bool outHalf, int act,
                               const float* __restrict__ post, long postLd) {
  const int tid = threadIdx.x;
  const int wid = tid >> 5, lane = tid & 31;
  const int lm = lane & 15, lg = lane >> 4;
  const int total = 4 * (N >> 4);
  for (int job = wid; job < total; job += WAVES) {
    int mr = job & 3, nc = job >> 2;
    v8f acc = {};
    int arow = mr * 16 + lm;
    int n = nc * 16 + lm;
    const half_t* wrow = W + (long)n * K;
    // prefetch next job's weight row (same lane pattern, n+32) into near caches
    const bool pf = (job + WAVES < total);
    const half_t* wnext = pf ? (W + (long)(n + 32) * K) : wrow;

    auto load_frag = [&](int kb, v16h& a, v16h& b) {
#pragma unroll
      for (int r = 0; r < 8; ++r) {
        int kp = (r < 4 ? 2 * r : 16 + 2 * (r - 4)) + 8 * lg;
        a[2 * r] = A[arow * lda + kb + kp];
        a[2 * r + 1] = A[arow * lda + kb + kp + 1];
        b[2 * r] = wrow[kb + kp];
        b[2 * r + 1] = wrow[kb + kp + 1];
      }
    };

    v16h a0, b0, a1, b1;
    load_frag(0, a0, b0);
    for (int kb = 0; kb < K; kb += 64) {
      if (pf) __builtin_prefetch(wnext + kb, 0, 3);
      if (kb + 32 < K) load_frag(kb + 32, a1, b1);
      acc = __builtin_amdgcn_wmma_f32_16x16x32_f16(false, a0, false, b0,
                                                   (short)0, acc, false, false);
      if (kb + 64 < K) load_frag(kb + 64, a0, b0);
      if (kb + 32 < K)
        acc = __builtin_amdgcn_wmma_f32_16x16x32_f16(false, a1, false, b1,
                                                     (short)0, acc, false, false);
    }

    float bb = bias ? bias[n] : 0.0f;
#pragma unroll
    for (int e = 0; e < 8; ++e) {
      int m = mr * 16 + lg * 8 + e;
      float p = (act == ACT_HALFSUM_TANH) ? post[(long)m * postLd + n] : 0.0f;
      float v = act_apply(acc[e] + bb, act, p);
      if (outHalf) ((half_t*)out)[m * ldo + n] = (half_t)v;
      else         ((float*)out)[m * ldo + n] = v;
    }
  }
  __syncthreads();
}

// ---------------- persistent GRU (40 steps, state in LDS) ----------------
__global__ __launch_bounds__(256) void gru_kernel(
    const float* __restrict__ gi,   // [64, 40, 1536] input gates (bias folded)
    const half_t* __restrict__ wh,  // [1536, 512] f16
    const float* __restrict__ bh,   // [1536]
    float* __restrict__ x_rnn) {    // [64, 40, 512]
  extern __shared__ char smem[];
  half_t* h = (half_t*)smem;        // [64][520]
  half_t* gh = h + 64 * 520;        // [64][1544]
  const int tid = threadIdx.x;
  for (int e = tid; e < 64 * 520; e += 256) h[e] = (half_t)0.0f;
  __syncthreads();

  for (int t = 0; t < 40; ++t) {
    block_gemm_lds(h, 520, 512, wh, bh, 1536, gh, 1544, true, ACT_NONE, nullptr, 0);
    for (int e = tid; e < 64 * 512; e += 256) {
      int b = e >> 9, n = e & 511;
      const float* gib = gi + ((long)(b * 40 + t)) * 1536;
      float ir = gib[n], iz = gib[512 + n], inn = gib[1024 + n];
      float hr = (float)gh[b * 1544 + n];
      float hz = (float)gh[b * 1544 + 512 + n];
      float hn = (float)gh[b * 1544 + 1024 + n];
      float r = fast_sigm(ir + hr);
      float z = fast_sigm(iz + hz);
      float nn = fast_tanh(inn + r * hn);
      float hold = (float)h[b * 520 + n];
      float hnew = (1.0f - z) * nn + z * hold;
      h[b * 520 + n] = (half_t)hnew;
      x_rnn[((long)(b * 40 + t)) * 512 + n] = hnew;
    }
    __syncthreads();
  }
}

// ---------------- persistent DKF latent scan (60 steps) ----------------
__global__ __launch_bounds__(256) void scan_kernel(
    const float* __restrict__ z_c,    // [64,60,64]
    const float* __restrict__ x_rnn,  // [64,40,512]
    const half_t* c1w, const float* c1b,
    const half_t* cmw, const float* cmb,
    const half_t* cvw, const float* cvb,
    const half_t* t1w, const float* t1b,
    const half_t* t2w, const float* t2b,
    const half_t* t3w, const float* t3b,
    const half_t* t4w, const float* t4b,
    const half_t* t0w, const float* t0b,
    const half_t* tvw, const float* tvb,
    half_t* __restrict__ z_seq,       // [64,60,64] f16 (feeds emission GEMM)
    float* __restrict__ out_muq, float* __restrict__ out_varq,
    float* __restrict__ out_mup, float* __restrict__ out_varp) {
  extern __shared__ char smem[];
  half_t* A0 = (half_t*)smem;        // [64][136]  (zz = [z | zc])
  half_t* A1 = A0 + 64 * 136;        // [64][520]  (hc / relu(t1,t3))
  float* MU = (float*)(A1 + 64 * 520);
  float* VAR = MU + 64 * 64;
  float* G = VAR + 64 * 64;
  float* HP = G + 64 * 64;
  float* MU0 = HP + 64 * 64;
  float* LV = MU0 + 64 * 64;
  float* Z = LV + 64 * 64;
  const int tid = threadIdx.x;
  for (int e = tid; e < 64 * 64; e += 256) Z[e] = 0.0f;
  __syncthreads();

  for (int t = 0; t < 60; ++t) {
    const bool obs = (t < 40);
    for (int e = tid; e < 64 * 64; e += 256) {
      int b = e >> 6, n = e & 63;
      A0[b * 136 + n] = (half_t)Z[e];
      A0[b * 136 + 64 + n] = (half_t)z_c[((long)b * 60 + t) * 64 + n];
    }
    __syncthreads();

    if (obs) {
      // hc = 0.5*(tanh(zz@c1^T+b) + x_rnn[:,t])
      block_gemm_lds(A0, 136, 128, c1w, c1b, 512, A1, 520, true,
                     ACT_HALFSUM_TANH, x_rnn + (long)t * 512, (long)40 * 512);
      block_gemm_lds(A1, 520, 512, cmw, cmb, 64, MU, 64, false, ACT_NONE, nullptr, 0);
      block_gemm_lds(A1, 520, 512, cvw, cvb, 64, VAR, 64, false, ACT_NONE, nullptr, 0);
      for (int e = tid; e < 64 * 64; e += 256) {
        int b = e >> 6, n = e & 63;
        float mu = MU[e], va = VAR[e];
        float eps = randn_at(0x2468ACEu, (unsigned)((t * 64 + b) * 64 + n));
        float zq = mu + eps * __expf(0.5f * va);
        Z[e] = zq;
        A0[b * 136 + n] = (half_t)zq;  // transition input
        long oi = ((long)b * 40 + t) * 64 + n;
        out_muq[oi] = mu;
        out_varq[oi] = va;
        if (t == 0) { out_mup[oi] = 0.0f; out_varp[oi] = 0.0f; }
        z_seq[((long)b * 60 + t) * 64 + n] = (half_t)zq;
      }
      __syncthreads();
    }

    // gated transition on [z | zc_t]
    block_gemm_lds(A0, 136, 128, t1w, t1b, 256, A1, 520, true, ACT_RELU, nullptr, 0);
    block_gemm_lds(A1, 520, 256, t2w, t2b, 64, G, 64, false, ACT_SIGM, nullptr, 0);
    block_gemm_lds(A0, 136, 128, t3w, t3b, 256, A1, 520, true, ACT_RELU, nullptr, 0);
    block_gemm_lds(A1, 520, 256, t4w, t4b, 64, HP, 64, false, ACT_NONE, nullptr, 0);
    block_gemm_lds(A0, 136, 64, t0w, t0b, 64, MU0, 64, false, ACT_NONE, nullptr, 0);
    for (int e = tid; e < 64 * 64; e += 256) {
      int b = e >> 6, n = e & 63;
      A0[b * 136 + n] = (half_t)fmaxf(HP[e], 0.0f);
    }
    __syncthreads();
    block_gemm_lds(A0, 136, 64, tvw, tvb, 64, LV, 64, false, ACT_NONE, nullptr, 0);

    for (int e = tid; e < 64 * 64; e += 256) {
      int b = e >> 6, n = e & 63;
      float g = G[e], hp = HP[e];
      float mu = (1.0f - g) * MU0[e] + g * hp;
      float lv = LV[e];
      if (obs) {
        if (t + 1 < 40) {
          long oi = ((long)b * 40 + (t + 1)) * 64 + n;
          out_mup[oi] = mu;
          out_varp[oi] = lv;
        }
      } else {
        float eps = randn_at(0x13579BDu, (unsigned)(((t - 40) * 64 + b) * 64 + n));
        float zp = mu + eps * __expf(0.5f * lv);
        Z[e] = zp;
        z_seq[((long)b * 60 + t) * 64 + n] = (half_t)zp;
      }
    }
    __syncthreads();
  }
}

// ---------------- small glue kernels ----------------

__global__ void convert_f32_to_f16(const float* __restrict__ s, half_t* __restrict__ d, long n) {
  long i = (long)blockIdx.x * 256 + threadIdx.x;
  if (i < n) d[i] = (half_t)s[i];
}

__global__ void mean_domain(const float* __restrict__ dlout, float* __restrict__ zc) {
  int i = blockIdx.x * 256 + threadIdx.x;  // B*T*64 = 245760
  if (i >= 64 * 60 * 64) return;
  int n = i & 63;
  int bt = i >> 6;
  int t = bt % 60, b = bt / 60;
  float s = 0.0f;
  for (int kd = 0; kd < 3; ++kd)
    s += dlout[((long)(b * 3 + kd) * 60 + t) * 64 + n];
  zc[i] = s * (1.0f / 3.0f);
}

__global__ void gather_xo(const float* __restrict__ x, half_t* __restrict__ xo) {
  long i = (long)blockIdx.x * 256 + threadIdx.x;  // 2560*1024
  if (i >= (long)2560 * 1024) return;
  long col = i & 1023;
  long row = i >> 10;
  long b = row / 40, t = row - b * 40;
  xo[i] = (half_t)x[((b * 60 + t) << 10) + col];
}

// ---------------- host orchestration ----------------

extern "C" void kernel_launch(void* const* d_in, const int* in_sizes, int n_in,
                              void* d_out, int out_size, void* d_ws, size_t ws_size,
                              hipStream_t stream) {
  (void)in_sizes; (void)n_in; (void)out_size; (void)ws_size;
  const float* x = (const float*)d_in[0];
  const float* D = (const float*)d_in[1];
  const float* dc1_w = (const float*)d_in[2];  const float* dc1_b = (const float*)d_in[3];
  const float* dc2_w = (const float*)d_in[4];  const float* dc2_b = (const float*)d_in[5];
  const float* dl_w = (const float*)d_in[6];   const float* dl_b = (const float*)d_in[7];
  const float* e1_w = (const float*)d_in[8];   const float* e1_b = (const float*)d_in[9];
  const float* e2_w = (const float*)d_in[10];  const float* e2_b = (const float*)d_in[11];
  const float* e3_w = (const float*)d_in[12];  const float* e3_b = (const float*)d_in[13];
  const float* gru_wi = (const float*)d_in[14]; const float* gru_bi = (const float*)d_in[15];
  const float* gru_wh = (const float*)d_in[16]; const float* gru_bh = (const float*)d_in[17];
  const float* t1_w = (const float*)d_in[18];  const float* t1_b = (const float*)d_in[19];
  const float* t2_w = (const float*)d_in[20];  const float* t2_b = (const float*)d_in[21];
  const float* t3_w = (const float*)d_in[22];  const float* t3_b = (const float*)d_in[23];
  const float* t4_w = (const float*)d_in[24];  const float* t4_b = (const float*)d_in[25];
  const float* t0_w = (const float*)d_in[26];  const float* t0_b = (const float*)d_in[27];
  const float* tv_w = (const float*)d_in[28];  const float* tv_b = (const float*)d_in[29];
  const float* c1_w = (const float*)d_in[30];  const float* c1_b = (const float*)d_in[31];
  const float* cm_w = (const float*)d_in[32];  const float* cm_b = (const float*)d_in[33];
  const float* cv_w = (const float*)d_in[34];  const float* cv_b = (const float*)d_in[35];
  const float* em_l_w = (const float*)d_in[36]; const float* em_l_b = (const float*)d_in[37];
  const float* em_d1_w = (const float*)d_in[38]; const float* em_d1_b = (const float*)d_in[39];
  const float* em_d2_w = (const float*)d_in[40]; const float* em_d2_b = (const float*)d_in[41];

  float* out = (float*)d_out;
  float* out_x = out;                    // [64,60,32,32]
  float* out_muq = out + 3932160;        // [64,40,64]
  float* out_varq = out + 4096000;
  float* out_mup = out + 4259840;
  float* out_varp = out + 4423680;

  char* wsp = (char*)d_ws;
  auto carve = [&](size_t bytes) -> void* {
    char* p = wsp;
    wsp += (bytes + 255) & ~(size_t)255;
    return (void*)p;
  };
  const long M1 = 11520L * 256;   // conv1 output pixels
  const long M2 = 11520L * 64;    // conv2 output pixels
  half_t* c1out = (half_t*)carve((size_t)M1 * 32 * 2);   // NHWC f16
  half_t* c2out = (half_t*)carve((size_t)M2 * 64 * 2);   // NHWC f16
  float* dlout = (float*)carve(11520L * 64 * 4);
  float* zc = (float*)carve(64L * 60 * 64 * 4);
  half_t* xo = (half_t*)carve(2560L * 1024 * 2);
  half_t* e1o = (half_t*)carve(2560L * 2048 * 2);
  half_t* e2o = (half_t*)carve(2560L * 2048 * 2);
  half_t* e3o = (half_t*)carve(2560L * 512 * 2);
  float* gi = (float*)carve(2560L * 1536 * 4);
  float* xrnn = (float*)carve(2560L * 512 * 4);
  half_t* zseq = (half_t*)carve(3840L * 64 * 2);
  half_t* emo = (half_t*)carve(3840L * 4096 * 2);        // NCHW rows
  half_t* d1o = (half_t*)carve(983040L * 32 * 2);        // NHWC f16
  // f16 weight copies (converted once per launch; stream from L2 thereafter)
  half_t* dc1w16 = (half_t*)carve(32L * 9 * 2);
  half_t* dc2w16 = (half_t*)carve(64L * 288 * 2);
  half_t* dlw16 = (half_t*)carve(64L * 4096 * 2);
  half_t* e1w16 = (half_t*)carve(2048L * 1024 * 2);
  half_t* e2w16 = (half_t*)carve(2048L * 2048 * 2);
  half_t* e3w16 = (half_t*)carve(512L * 2048 * 2);
  half_t* gwi16 = (half_t*)carve(1536L * 512 * 2);
  half_t* emlw16 = (half_t*)carve(4096L * 64 * 2);
  half_t* ed1w16 = (half_t*)carve(32L * 1024 * 2);
  half_t* ed2w16 = (half_t*)carve(1L * 512 * 2);
  half_t* wh16 = (half_t*)carve(1536L * 512 * 2);
  half_t* c1w16 = (half_t*)carve(512L * 128 * 2);
  half_t* cmw16 = (half_t*)carve(64L * 512 * 2);
  half_t* cvw16 = (half_t*)carve(64L * 512 * 2);
  half_t* t1w16 = (half_t*)carve(256L * 128 * 2);
  half_t* t2w16 = (half_t*)carve(64L * 256 * 2);
  half_t* t3w16 = (half_t*)carve(256L * 128 * 2);
  half_t* t4w16 = (half_t*)carve(64L * 256 * 2);
  half_t* t0w16 = (half_t*)carve(64L * 64 * 2);
  half_t* tvw16 = (half_t*)carve(64L * 64 * 2);

  ConvGeom gC1 = {1, 32, 32, 16, 16, 3, 3, 2, 1, 0, 0};
  ConvGeom gC2 = {32, 16, 16, 8, 8, 3, 3, 2, 1, 0, 0};
  ConvGeom gD1 = {64, 8, 8, 16, 16, 4, 4, 1, 2, 1, 1};    // input NCHW rows
  ConvGeom gD2 = {32, 16, 16, 32, 32, 4, 4, 1, 2, 1, 0};  // input NHWC
  ConvGeom gNull = {};

  dim3 blk(256);
  auto tohalf = [&](const float* s, half_t* d, long n) {
    convert_f32_to_f16<<<dim3((unsigned)((n + 255) / 256)), blk, 0, stream>>>(s, d, n);
  };
  tohalf(dc1_w, dc1w16, 32L * 9);
  tohalf(dc2_w, dc2w16, 64L * 288);
  tohalf(dl_w, dlw16, 64L * 4096);
  tohalf(e1_w, e1w16, 2048L * 1024);
  tohalf(e2_w, e2w16, 2048L * 2048);
  tohalf(e3_w, e3w16, 512L * 2048);
  tohalf(gru_wi, gwi16, 1536L * 512);
  tohalf(em_l_w, emlw16, 4096L * 64);
  tohalf(em_d1_w, ed1w16, 32L * 1024);
  tohalf(em_d2_w, ed2w16, 1L * 512);
  tohalf(gru_wh, wh16, 1536L * 512);
  tohalf(c1_w, c1w16, 512L * 128);
  tohalf(cm_w, cmw16, 64L * 512);
  tohalf(cv_w, cvw16, 64L * 512);
  tohalf(t1_w, t1w16, 256L * 128);
  tohalf(t2_w, t2w16, 64L * 256);
  tohalf(t3_w, t3w16, 256L * 128);
  tohalf(t4_w, t4w16, 64L * 256);
  tohalf(t0_w, t0w16, 64L * 64);
  tohalf(tv_w, tvw16, 64L * 64);

  // domain CNN (implicit im2col WMMA GEMMs)
  gemm_tiled<2, false, true, false><<<dim3(23040, 1), blk, 0, stream>>>(
      D, dc1w16, dc1_b, c1out, M1, 32, 9, ACT_RELU, 0, 0, gC1);
  gemm_tiled<2, true, true, false><<<dim3(5760, 1), blk, 0, stream>>>(
      c1out, dc2w16, dc2_b, c2out, M2, 64, 288, ACT_RELU, 0, 0, gC2);
  gemm_tiled<1, true, false, false><<<dim3(90, 1), blk, 0, stream>>>(
      c2out, dlw16, dl_b, dlout, 11520L, 64, 4096, ACT_NONE, 64, 64, gNull);
  mean_domain<<<dim3((245760 + 255) / 256), blk, 0, stream>>>(dlout, zc);

  // embedding MLP + GRU input gates (batched; async DMA staged)
  gather_xo<<<dim3((2621440 + 255) / 256), blk, 0, stream>>>(x, xo);
  gemm_tiled<0, true, true, true><<<dim3(20, 32), blk, 0, stream>>>(
      xo, e1w16, e1_b, e1o, 2560L, 2048, 1024, ACT_RELU, 0, 0, gNull);
  gemm_tiled<0, true, true, true><<<dim3(20, 32), blk, 0, stream>>>(
      e1o, e2w16, e2_b, e2o, 2560L, 2048, 2048, ACT_RELU, 0, 0, gNull);
  gemm_tiled<0, true, true, true><<<dim3(20, 8), blk, 0, stream>>>(
      e2o, e3w16, e3_b, e3o, 2560L, 512, 2048, ACT_RELU, 0, 0, gNull);
  gemm_tiled<0, true, false, true><<<dim3(20, 24), blk, 0, stream>>>(
      e3o, gwi16, gru_bi, gi, 2560L, 1536, 512, ACT_NONE, 0, 0, gNull);

  // persistent recurrences (single block, state in LDS)
  size_t gruSmem = (size_t)(64 * 520 + 64 * 1544) * sizeof(half_t);    // 264 KB
  gru_kernel<<<dim3(1), blk, gruSmem, stream>>>(gi, wh16, gru_bh, xrnn);

  size_t scanSmem = (size_t)(64 * 136 + 64 * 520) * sizeof(half_t) +
                    (size_t)7 * 64 * 64 * sizeof(float);               // ~199 KB
  scan_kernel<<<dim3(1), blk, scanSmem, stream>>>(
      zc, xrnn, c1w16, c1_b, cmw16, cm_b, cvw16, cv_b,
      t1w16, t1_b, t2w16, t2_b, t3w16, t3_b, t4w16, t4_b,
      t0w16, t0_b, tvw16, tv_b,
      zseq, out_muq, out_varq, out_mup, out_varp);

  // emission decoder
  gemm_tiled<0, true, true, true><<<dim3(30, 64), blk, 0, stream>>>(
      zseq, emlw16, em_l_b, emo, 3840L, 4096, 64, ACT_RELU, 0, 0, gNull);
  gemm_tiled<2, true, true, false><<<dim3(7680, 1), blk, 0, stream>>>(
      emo, ed1w16, em_d1_b, d1o, 983040L, 32, 1024, ACT_RELU, 0, 0, gD1);
  gemm_tiled<2, true, false, false><<<dim3(30720, 1), blk, 0, stream>>>(
      d1o, ed2w16, em_d2_b, out_x, 3932160L, 1, 512, ACT_NONE, 0, 0, gD2);
}
